// RetNet_87007447482897
// MI455X (gfx1250) — compile-verified, matrix operand order checked by reading, not run on hardware
//
#include <hip/hip_runtime.h>
#include <hip/hip_bf16.h>
#include <math.h>

#define S_  2048
#define D_  512
#define H_  8
#define DH_ 64
#define L_  4
#define FF_ 2048

typedef __attribute__((ext_vector_type(16))) __bf16 v16bf;
typedef __attribute__((ext_vector_type(8)))  float  v8f;
typedef __attribute__((ext_vector_type(4)))  int    v4i;

#if defined(__HIP_DEVICE_COMPILE__) && \
    __has_builtin(__builtin_amdgcn_global_load_async_to_lds_b128) && \
    __has_builtin(__builtin_amdgcn_s_wait_asynccnt)
#define HAS_ASYNC_LDS 1
#else
#define HAS_ASYNC_LDS 0
#endif

__device__ __forceinline__ __bf16 f2bf(float f) { return (__bf16)f; }

__device__ __forceinline__ v8f wmma_bf16(v16bf a, v16bf b, v8f c) {
  // D = A(16x32 bf16) * B(32x16 bf16) + C(16x16 f32)
  return __builtin_amdgcn_wmma_f32_16x16x32_bf16(false, a, false, b, (short)0, c, false, false);
}

// ---------------------------------------------------------------------------
// x_ws = x + pos_emb[0,:]  (reference broadcasts pos_emb[:B] == row 0 over S)
// ---------------------------------------------------------------------------
__global__ void add_pos_k(const float* __restrict__ x, const float* __restrict__ pe,
                          float* __restrict__ xw) {
  int i = blockIdx.x * 256 + threadIdx.x;
  if (i < S_ * D_) xw[i] = x[i] + pe[i % D_];
}

// ---------------------------------------------------------------------------
// Convert W[K][N] f32 (row major) -> Wt[N][K] bf16 (B-operand friendly)
// ---------------------------------------------------------------------------
__global__ void convT_k(const float* __restrict__ W, __bf16* __restrict__ Wt,
                        int K, int N) {
  int idx = blockIdx.x * 256 + threadIdx.x;
  if (idx >= K * N) return;
  int k = idx / N, n = idx % N;
  Wt[(size_t)n * K + k] = f2bf(W[idx]);
}

// ---------------------------------------------------------------------------
// LayerNorm over D, affine, output bf16 (GEMM A operand)
// ---------------------------------------------------------------------------
__global__ void ln_k(const float* __restrict__ x, const float* __restrict__ w,
                     const float* __restrict__ b, __bf16* __restrict__ out) {
  int s = blockIdx.x, t = threadIdx.x;           // 128 threads
  __shared__ float r1[128], r2[128];
  float sum = 0.f, sq = 0.f;
  for (int i = t; i < D_; i += 128) { float v = x[s * D_ + i]; sum += v; sq += v * v; }
  r1[t] = sum; r2[t] = sq; __syncthreads();
  for (int st = 64; st > 0; st >>= 1) {
    if (t < st) { r1[t] += r1[t + st]; r2[t] += r2[t + st]; }
    __syncthreads();
  }
  float mu = r1[0] / D_;
  float var = r2[0] / D_ - mu * mu;
  float rs = rsqrtf(var + 1e-5f);
  for (int i = t; i < D_; i += 128)
    out[s * D_ + i] = f2bf((x[s * D_ + i] - mu) * rs * w[i] + b[i]);
}

// ---------------------------------------------------------------------------
// theta-shift (RoPE-style) on q (scaled 1/sqrt(DH)) and k -> bf16;
// v -> per-head transposed bf16 vt[h*DH+d][s] (B operand of sim@v)
// ---------------------------------------------------------------------------
__global__ void rope_k(const float* __restrict__ q, const float* __restrict__ k,
                       const float* __restrict__ v, __bf16* __restrict__ qb,
                       __bf16* __restrict__ kb, __bf16* __restrict__ vt) {
  int idx = blockIdx.x * 256 + threadIdx.x;      // over S*D/2 pairs
  if (idx >= S_ * D_ / 2) return;
  int s = idx / (D_ / 2);
  int rem = idx % (D_ / 2);
  int h = rem / (DH_ / 2);
  int j = rem % (DH_ / 2);
  float theta = __powf(10000.f, -(2.f * (float)j) / (float)DH_);
  float ang = (float)s * theta;
  float sn, cs;
  __sincosf(ang, &sn, &cs);
  int base = s * D_ + h * DH_ + 2 * j;
  float q0 = q[base], q1 = q[base + 1];
  qb[base]     = f2bf((q0 * cs - q1 * sn) * 0.125f);   // 1/sqrt(64)
  qb[base + 1] = f2bf((q1 * cs + q0 * sn) * 0.125f);
  float k0 = k[base], k1 = k[base + 1];
  kb[base]     = f2bf(k0 * cs - k1 * sn);
  kb[base + 1] = f2bf(k1 * cs + k0 * sn);
  int dh = h * DH_ + 2 * j;
  vt[(size_t)dh * S_ + s]       = f2bf(v[base]);
  vt[(size_t)(dh + 1) * S_ + s] = f2bf(v[base + 1]);
}

// ---------------------------------------------------------------------------
// Retention: ret[16 query rows x DH] per (row-tile, head).
// sim = (q kT) * gamma^(n-m), causal; ret = sim @ v.  One wave per block.
// sim tile goes C-layout -> LDS(bf16) -> A-layout for the second WMMA.
// Decay weights: one expf per lane per tile, then running multiply by gamma.
// ---------------------------------------------------------------------------
__global__ void retention_k(const __bf16* __restrict__ qb, const __bf16* __restrict__ kb,
                            const __bf16* __restrict__ vt, float* __restrict__ ret) {
  __shared__ __bf16 simlds[16 * 32];
  int rowBase = blockIdx.x * 16;
  int h = blockIdx.y;
  int lane = threadIdx.x;
  int r = lane & 15, hf = lane >> 4;

  // A operand (q rows, K = DH split into 2 chunks of 32) — invariant over m loop
  const __bf16* qrow = qb + (size_t)(rowBase + r) * D_ + h * DH_ + hf * 16;
  v16bf a0 = *(const v16bf*)(qrow);
  v16bf a1 = *(const v16bf*)(qrow + 32);

  v8f rc[4] = {};   // 16 x 64 output (4 C tiles)

  float t0 = logf(1.f / 32.f), t1 = logf(1.f / 512.f);
  float gamma = 1.f - __expf(t0 + (float)h * (t1 - t0) / (float)(H_ - 1));
  float lg = logf(gamma);

  int nSteps = rowBase / 32 + 1;                 // causal coverage in 32-key steps
  for (int stp = 0; stp < nSteps; ++stp) {
    int mBase = stp * 32;
#pragma unroll
    for (int t = 0; t < 2; ++t) {
      int key = mBase + t * 16 + r;              // B operand: lane's N = key row
      const __bf16* krow = kb + (size_t)key * D_ + h * DH_ + hf * 16;
      v16bf b0 = *(const v16bf*)(krow);
      v16bf b1 = *(const v16bf*)(krow + 32);
      v8f z = {};
      v8f sacc = wmma_bf16(a0, b0, z);
      sacc = wmma_bf16(a1, b1, sacc);
      // decay mask in C-layout: M = i + 8*hf, N = r. dist(i) = d0 + i.
      int d0 = rowBase + 8 * hf - key;           // dist at i=0, d0 >= -31
      float w = __expf(lg * (float)d0);          // gamma^d0 (<= ~2.7, no ovfl)
#pragma unroll
      for (int i = 0; i < 8; ++i) {
        float wm = ((d0 + i) >= 0) ? w : 0.f;
        simlds[(i + 8 * hf) * 32 + t * 16 + r] = f2bf(sacc[i] * wm);
        w *= gamma;
      }
    }
    __syncthreads();
    // reload sim tile as A operand: row = lane&15, K chunk = lane>>4
    v16bf sa = *(const v16bf*)(&simlds[r * 32 + hf * 16]);
    __syncthreads();
#pragma unroll
    for (int j = 0; j < 4; ++j) {
      const __bf16* vrow = vt + (size_t)(h * DH_ + j * 16 + r) * S_ + mBase + hf * 16;
      v16bf bv = *(const v16bf*)(vrow);
      rc[j] = wmma_bf16(sa, bv, rc[j]);
    }
  }
  // store f32 result, C-layout
#pragma unroll
  for (int j = 0; j < 4; ++j)
#pragma unroll
    for (int i = 0; i < 8; ++i) {
      int row = rowBase + i + 8 * hf;
      ret[(size_t)row * D_ + h * DH_ + j * 16 + r] = rc[j][i];
    }
}

// ---------------------------------------------------------------------------
// GroupNorm over DH (no affine, eps 1e-6) then gate with silu(g), bf16 out
// ---------------------------------------------------------------------------
__global__ void gn_gate_k(const float* __restrict__ ret, const float* __restrict__ g,
                          __bf16* __restrict__ out) {
  int s = blockIdx.x, h = blockIdx.y, t = threadIdx.x;   // 64 threads
  __shared__ float r1[64], r2[64];
  int base = s * D_ + h * DH_;
  float v = ret[base + t];
  r1[t] = v; r2[t] = v * v; __syncthreads();
  for (int st = 32; st > 0; st >>= 1) {
    if (t < st) { r1[t] += r1[t + st]; r2[t] += r2[t + st]; }
    __syncthreads();
  }
  float mu = r1[0] / DH_;
  float var = r2[0] / DH_ - mu * mu;
  float rs = rsqrtf(var + 1e-6f);
  float gg = g[base + t];
  out[base + t] = f2bf((gg / (1.f + __expf(-gg))) * (v - mu) * rs);
}

// ---------------------------------------------------------------------------
// WMMA GEMM: C[M,N] = A_bf16[M,K] @ Bt_bf16[N,K]^T + bias
// MODE 0: f32 out, optional residual add (addsrc). MODE 1: silu -> bf16 out.
// Block = 128 threads (4 waves); wave owns a 16x64 tile.
// Shared A tile (16x32) staged once per block into LDS via async-to-LDS
// (double buffered, ASYNCcnt-tracked); B streamed from global per wave.
// ---------------------------------------------------------------------------
template <int MODE>
__global__ void gemm16_k(const __bf16* __restrict__ A, const __bf16* __restrict__ Bt,
                         const float* __restrict__ bias, void* __restrict__ Cout,
                         const float* __restrict__ addsrc, int K, int N) {
  __shared__ __bf16 atile[2][16 * 32];           // 2 x 1 KiB double buffer
  int tid = threadIdx.x;
  int lane = tid & 31, wave = tid >> 5;
  int rowBase = blockIdx.x * 16;
  int colBase = blockIdx.y * 256 + wave * 64;
  int r = lane & 15, hf = lane >> 4;

  const __bf16* brow = Bt + (size_t)(colBase + r) * K + hf * 16;
  size_t bstride = (size_t)16 * K;
  v8f acc[4] = {};

  // staging: 64 threads each move 16 bytes (8 bf16) of the A tile
  int srow = tid >> 2, sseg = tid & 3;
  const __bf16* asrc = A + (size_t)(rowBase + srow) * K + sseg * 8;

  if (tid < 64) {
#if HAS_ASYNC_LDS
    __builtin_amdgcn_global_load_async_to_lds_b128(
        (__attribute__((address_space(1))) v4i*)(asrc),
        (__attribute__((address_space(3))) v4i*)(&atile[0][srow * 32 + sseg * 8]),
        0, 0);
#else
    *(float4*)(&atile[0][srow * 32 + sseg * 8]) = *(const float4*)(asrc);
#endif
  }

  int bb = 0;
  for (int k0 = 0; k0 < K; k0 += 32, bb ^= 1) {
    bool more = (k0 + 32) < K;
    if (more && tid < 64) {                      // prefetch next buffer
#if HAS_ASYNC_LDS
      __builtin_amdgcn_global_load_async_to_lds_b128(
          (__attribute__((address_space(1))) v4i*)(asrc + k0 + 32),
          (__attribute__((address_space(3))) v4i*)(&atile[bb ^ 1][srow * 32 + sseg * 8]),
          0, 0);
#else
      *(float4*)(&atile[bb ^ 1][srow * 32 + sseg * 8]) = *(const float4*)(asrc + k0 + 32);
#endif
    }
#if HAS_ASYNC_LDS
    if (more) __builtin_amdgcn_s_wait_asynccnt(1);   // older (current) buf done
    else      __builtin_amdgcn_s_wait_asynccnt(0);
#endif
    __syncthreads();                             // current buffer visible to all
    v16bf a  = *(const v16bf*)(&atile[bb][r * 32 + hf * 16]);
    v16bf b0 = *(const v16bf*)(brow + k0);
    v16bf b1 = *(const v16bf*)(brow + k0 + bstride);
    v16bf b2 = *(const v16bf*)(brow + k0 + 2 * bstride);
    v16bf b3 = *(const v16bf*)(brow + k0 + 3 * bstride);
    acc[0] = wmma_bf16(a, b0, acc[0]);
    acc[1] = wmma_bf16(a, b1, acc[1]);
    acc[2] = wmma_bf16(a, b2, acc[2]);
    acc[3] = wmma_bf16(a, b3, acc[3]);
    __syncthreads();                             // reads done before re-stage
  }

#pragma unroll
  for (int j = 0; j < 4; ++j) {
    int col = colBase + j * 16 + r;
    float bb2 = bias[col];
#pragma unroll
    for (int i = 0; i < 8; ++i) {
      int row = rowBase + i + 8 * hf;
      float v = acc[j][i] + bb2;
      if (MODE == 0) {
        if (addsrc) v += addsrc[(size_t)row * N + col];
        ((float*)Cout)[(size_t)row * N + col] = v;
      } else {
        ((__bf16*)Cout)[(size_t)row * N + col] = f2bf(v / (1.f + __expf(-v)));
      }
    }
  }
}

// ---------------------------------------------------------------------------
// Final head: out[s] = sigmoid(x[s,:] . Wout + bout)
// ---------------------------------------------------------------------------
__global__ void head_k(const float* __restrict__ x, const float* __restrict__ Wout,
                       const float* __restrict__ bout, float* __restrict__ out) {
  int s = blockIdx.x, t = threadIdx.x;           // 128 threads
  __shared__ float r1[128];
  float sum = 0.f;
  for (int i = t; i < D_; i += 128) sum += x[s * D_ + i] * Wout[i];
  r1[t] = sum; __syncthreads();
  for (int st = 64; st > 0; st >>= 1) {
    if (t < st) r1[t] += r1[t + st];
    __syncthreads();
  }
  if (t == 0) out[s] = 1.f / (1.f + __expf(-(r1[0] + bout[0])));
}

// ---------------------------------------------------------------------------
extern "C" void kernel_launch(void* const* d_in, const int* in_sizes, int n_in,
                              void* d_out, int out_size, void* d_ws, size_t ws_size,
                              hipStream_t stream) {
  const float* x    = (const float*)d_in[0];
  const float* pe   = (const float*)d_in[1];
  const float* Wq   = (const float*)d_in[2];
  const float* bq   = (const float*)d_in[3];
  const float* Wk   = (const float*)d_in[4];
  const float* bk   = (const float*)d_in[5];
  const float* Wv   = (const float*)d_in[6];
  const float* bv   = (const float*)d_in[7];
  const float* Wg   = (const float*)d_in[8];
  const float* bg   = (const float*)d_in[9];
  const float* Wo   = (const float*)d_in[10];
  const float* bo   = (const float*)d_in[11];
  const float* ln1w = (const float*)d_in[12];
  const float* ln1b = (const float*)d_in[13];
  const float* ln2w = (const float*)d_in[14];
  const float* ln2b = (const float*)d_in[15];
  const float* W1   = (const float*)d_in[16];
  const float* b1   = (const float*)d_in[17];
  const float* W2   = (const float*)d_in[18];
  const float* b2   = (const float*)d_in[19];
  const float* Wout = (const float*)d_in[20];
  const float* bout = (const float*)d_in[21];

  char* base = (char*)d_ws;
  size_t off = 0;
  auto take = [&](size_t bytes) -> char* {
    off = (off + 255) & ~(size_t)255;
    char* p = base + off;
    off += bytes;
    return p;
  };

  const size_t SDf = (size_t)S_ * D_ * sizeof(float);
  const size_t SDb = (size_t)S_ * D_ * sizeof(__bf16);

  float*  xw  = (float*)take(SDf);
  float*  qf  = (float*)take(SDf);
  float*  kf  = (float*)take(SDf);
  float*  vf  = (float*)take(SDf);
  float*  gf  = (float*)take(SDf);
  float*  rt  = (float*)take(SDf);
  __bf16* ybf = (__bf16*)take(SDb);
  __bf16* qb  = (__bf16*)take(SDb);
  __bf16* kb  = (__bf16*)take(SDb);
  __bf16* vtb = (__bf16*)take(SDb);
  __bf16* gat = (__bf16*)take(SDb);
  __bf16* zbf = (__bf16*)take(SDb);
  __bf16* f1b = (__bf16*)take((size_t)S_ * FF_ * sizeof(__bf16));

  __bf16 *WqT[L_], *WkT[L_], *WvT[L_], *WgT[L_], *WoT[L_], *W1T[L_], *W2T[L_];
  for (int l = 0; l < L_; ++l) {
    WqT[l] = (__bf16*)take((size_t)D_ * D_ * 2);
    WkT[l] = (__bf16*)take((size_t)D_ * D_ * 2);
    WvT[l] = (__bf16*)take((size_t)D_ * D_ * 2);
    WgT[l] = (__bf16*)take((size_t)D_ * D_ * 2);
    WoT[l] = (__bf16*)take((size_t)D_ * D_ * 2);
    W1T[l] = (__bf16*)take((size_t)FF_ * D_ * 2);
    W2T[l] = (__bf16*)take((size_t)D_ * FF_ * 2);
  }

  // ---- weight prep: f32 [K][N] -> bf16 [N][K] ----
  int gDD = (D_ * D_ + 255) / 256;
  int gDF = (D_ * FF_ + 255) / 256;
  for (int l = 0; l < L_; ++l) {
    size_t oDD = (size_t)l * D_ * D_;
    size_t oDF = (size_t)l * D_ * FF_;
    convT_k<<<gDD, 256, 0, stream>>>(Wq + oDD, WqT[l], D_, D_);
    convT_k<<<gDD, 256, 0, stream>>>(Wk + oDD, WkT[l], D_, D_);
    convT_k<<<gDD, 256, 0, stream>>>(Wv + oDD, WvT[l], D_, D_);
    convT_k<<<gDD, 256, 0, stream>>>(Wg + oDD, WgT[l], D_, D_);
    convT_k<<<gDD, 256, 0, stream>>>(Wo + oDD, WoT[l], D_, D_);
    convT_k<<<gDF, 256, 0, stream>>>(W1 + oDF, W1T[l], D_, FF_);
    convT_k<<<gDF, 256, 0, stream>>>(W2 + oDF, W2T[l], FF_, D_);
  }

  // ---- x = x + pos_emb broadcast ----
  add_pos_k<<<(S_ * D_ + 255) / 256, 256, 0, stream>>>(x, pe, xw);

  dim3 gP(S_ / 16, D_ / 256);    // N = 512 GEMMs
  dim3 gF(S_ / 16, FF_ / 256);   // N = 2048 GEMM

  for (int l = 0; l < L_; ++l) {
    const float* bql = bq + (size_t)l * D_;
    const float* bkl = bk + (size_t)l * D_;
    const float* bvl = bv + (size_t)l * D_;
    const float* bgl = bg + (size_t)l * D_;
    const float* bol = bo + (size_t)l * D_;
    const float* b1l = b1 + (size_t)l * FF_;
    const float* b2l = b2 + (size_t)l * D_;

    // y = LN1(x) -> bf16
    ln_k<<<S_, 128, 0, stream>>>(xw, ln1w + (size_t)l * D_, ln1b + (size_t)l * D_, ybf);

    // q,k,v,g projections (f32 out)
    gemm16_k<0><<<gP, 128, 0, stream>>>(ybf, WqT[l], bql, (void*)qf, nullptr, D_, D_);
    gemm16_k<0><<<gP, 128, 0, stream>>>(ybf, WkT[l], bkl, (void*)kf, nullptr, D_, D_);
    gemm16_k<0><<<gP, 128, 0, stream>>>(ybf, WvT[l], bvl, (void*)vf, nullptr, D_, D_);
    gemm16_k<0><<<gP, 128, 0, stream>>>(ybf, WgT[l], bgl, (void*)gf, nullptr, D_, D_);

    // theta shift + scale + v transpose -> bf16 operands
    rope_k<<<(S_ * D_ / 2 + 255) / 256, 256, 0, stream>>>(qf, kf, vf, qb, kb, vtb);

    // retention (causal decay-masked attention) -> f32
    retention_k<<<dim3(S_ / 16, H_), 32, 0, stream>>>(qb, kb, vtb, rt);

    // groupnorm + silu(g) gate -> bf16
    gn_gate_k<<<dim3(S_, H_), 64, 0, stream>>>(rt, gf, gat);

    // x = x + gated @ Wo + bo (residual fused)
    gemm16_k<0><<<gP, 128, 0, stream>>>(gat, WoT[l], bol, (void*)xw, xw, D_, D_);

    // z = LN2(x) -> bf16
    ln_k<<<S_, 128, 0, stream>>>(xw, ln2w + (size_t)l * D_, ln2b + (size_t)l * D_, zbf);

    // f1 = silu(z @ W1 + b1) -> bf16
    gemm16_k<1><<<gF, 128, 0, stream>>>(zbf, W1T[l], b1l, (void*)f1b, nullptr, D_, FF_);

    // x = x + f1 @ W2 + b2 (residual fused)
    gemm16_k<0><<<gP, 128, 0, stream>>>(f1b, W2T[l], b2l, (void*)xw, xw, FF_, D_);
  }

  // out = sigmoid(x @ Wout + bout)
  head_k<<<S_, 128, 0, stream>>>(xw, Wout, bout, (float*)d_out);
}
